// MGCN_69406671503638
// MI455X (gfx1250) — matrix operand order repaired and checked by reading, not compiled, not run on hardware
//
#include <hip/hip_runtime.h>
#include <math.h>
#include <stdint.h>

// ---------------- problem constants (match reference) ----------------
#define NN_ 4096      // nodes
#define FF_ 128       // feature dim
#define HH_ 128       // hidden dim
#define CC_ 10        // classes
#define KG_ 4         // graph blocks
#define GH_ 512       // 4*H (LSTM gate dim)

typedef __bf16 bf16_t;
typedef __attribute__((ext_vector_type(16))) __bf16 v16bf;
typedef __attribute__((ext_vector_type(8)))  float  v8f;
typedef __attribute__((ext_vector_type(4)))  unsigned int v4u;

static __device__ __forceinline__ bf16_t f2bf(float x) { return (bf16_t)x; }
static __device__ __forceinline__ float sigm(float x) { return 1.0f / (1.0f + __expf(-x)); }

// -------- async Global->LDS staging (gfx1250 path, ASYNCcnt-tracked) --------
#if defined(__has_builtin)
# if __has_builtin(__builtin_amdgcn_global_load_async_to_lds_b128) && \
     __has_builtin(__builtin_amdgcn_s_wait_asynccnt)
#  define USE_ASYNC_LDS 1
# endif
#endif

typedef int v4i_vs __attribute__((vector_size(16)));
typedef __attribute__((address_space(1))) v4i_vs* gas1_v4i;
typedef __attribute__((address_space(3))) v4i_vs* gas3_v4i;

static __device__ __forceinline__ void stage16(const bf16_t* g, bf16_t* l)
{
#ifdef USE_ASYNC_LDS
    // LDS generic address: low 32 bits are the LDS byte offset (ISA 10.2).
    __builtin_amdgcn_global_load_async_to_lds_b128(
        (gas1_v4i)(uintptr_t)g,
        (gas3_v4i)(uint32_t)(uintptr_t)l,
        0, 0);
#else
    *(v4u*)l = *(const v4u*)g;
#endif
}

static __device__ __forceinline__ void stage_wait()
{
#ifdef USE_ASYNC_LDS
    __builtin_amdgcn_s_wait_asynccnt(0);
#endif
    __syncthreads();
}

// =====================================================================
// WMMA GEMM with LDS-staged B panel:
//   C[M,N] (fp32) = A[M,Kd](bf16,row-major,lda) @ Bt[N,Kd](bf16,row-major,ldb)^T
//   (+ bias[N]) (+= if accumulate)
// block = (32,8): 8 waves; block tile = 128(M) x 64(N).
// Each wave owns a 16-row strip and computes 4 WMMA tiles (16x64) per K-step.
// B panel (64 rows x 32 K = 4KB) double-buffered in LDS via async loads.
// grid = (N/64, M/128). Requires M%128==0, N%64==0, Kd%32==0.
// =====================================================================
__launch_bounds__(256)
__global__ void gemm_wmma_stage(const bf16_t* __restrict__ A, int lda,
                                const bf16_t* __restrict__ Bt, int ldb,
                                float* __restrict__ C, int ldc,
                                int Kd, const float* __restrict__ bias,
                                int accumulate)
{
    __shared__ bf16_t bsh[2][64 * 32];          // 2 x 4KB double buffer

    const int lane = threadIdx.x;               // 0..31 (wave32)
    const int wave = threadIdx.y;               // 0..7
    const int half = lane >> 4;                 // 0 or 1
    const int lr   = lane & 15;
    const int m0 = blockIdx.y * 128 + wave * 16;
    const int n0 = blockIdx.x * 64;
    const int tid = wave * 32 + lane;           // 0..255

    // staging: 64 rows x 64B; each thread moves 16B. row = tid/4, elem = (tid%4)*8
    const int srow  = tid >> 2;
    const int selem = (tid & 3) * 8;
    const bf16_t* bsrc = Bt + (size_t)(n0 + srow) * ldb + selem;
    bf16_t* bdst[2] = { &bsh[0][srow * 32 + selem], &bsh[1][srow * 32 + selem] };

    // A fragment source: lane<16 row(m0+lr) K{0..7,16..23}; lane>=16 K{8..15,24..31}
    const bf16_t* arow = A + (size_t)(m0 + lr) * lda + half * 8;

    v8f acc[4] = { {}, {}, {}, {} };

    stage16(bsrc, bdst[0]);                     // preload K-panel 0
    stage_wait();

    for (int k0 = 0; k0 < Kd; k0 += 32) {
        const int cur = (k0 >> 5) & 1;
        if (k0 + 32 < Kd) {
            stage16(bsrc + k0 + 32, bdst[cur ^ 1]);   // async prefetch next panel
            __builtin_prefetch(arow + k0 + 32, 0, 0); // global_prefetch_b8 for A
        }
        union { v16bf v; v4u u[2]; } a;
        a.u[0] = *(const v4u*)(arow + k0);
        a.u[1] = *(const v4u*)(arow + k0 + 16);
        const bf16_t* bb = &bsh[cur][0];
        #pragma unroll
        for (int tN = 0; tN < 4; ++tN) {
            // Bt row (n0 + tN*16 + lr): lanes<16 K{0..7,16..23}, lanes>=16 K{8..15,24..31}
            const bf16_t* bl = bb + (tN * 16 + lr) * 32 + half * 8;
            union { v16bf v; v4u u[2]; } b;
            b.u[0] = *(const v4u*)(bl);         // ds_load_b128
            b.u[1] = *(const v4u*)(bl + 16);
            acc[tN] = __builtin_amdgcn_wmma_f32_16x16x32_bf16(
                          false, a.v, false, b.v, (short)0, acc[tN], false, false);
        }
        stage_wait();                           // async done + all reads of cur done
    }

    // C/D layout: lane<16 -> rows m0+0..7, lane>=16 -> rows m0+8..15; col n-tile + lr
    #pragma unroll
    for (int tN = 0; tN < 4; ++tN) {
        const int col = n0 + tN * 16 + lr;
        float* crow = C + (size_t)(m0 + half * 8) * ldc + col;
        const float bv = bias ? bias[col] : 0.0f;
        if (accumulate) {
            #pragma unroll
            for (int j = 0; j < 8; ++j) crow[(size_t)j * ldc] += acc[tN][j];
        } else {
            #pragma unroll
            for (int j = 0; j < 8; ++j) crow[(size_t)j * ldc] = acc[tN][j] + bv;
        }
    }
}

// ------------- degree: dinv[k][i] = rsqrt(1 + sum_j A[i,j,k]) -------------
__global__ void degnorm_kernel(const float4* __restrict__ adj, float* __restrict__ dinv)
{
    __shared__ float4 red[256];
    const int i = blockIdx.x;
    float4 s = make_float4(0.f, 0.f, 0.f, 0.f);
    for (int j = threadIdx.x; j < NN_; j += 256) {
        float4 a = adj[(size_t)i * NN_ + j];
        s.x += a.x; s.y += a.y; s.z += a.z; s.w += a.w;
    }
    red[threadIdx.x] = s;
    __syncthreads();
    for (int off = 128; off > 0; off >>= 1) {
        if (threadIdx.x < off) {
            float4 o = red[threadIdx.x + off];
            red[threadIdx.x].x += o.x; red[threadIdx.x].y += o.y;
            red[threadIdx.x].z += o.z; red[threadIdx.x].w += o.w;
        }
        __syncthreads();
    }
    if (threadIdx.x < 4) {
        float d = 1.0f + ((const float*)&red[0])[threadIdx.x];
        dinv[threadIdx.x * NN_ + i] = rsqrtf(d);
    }
}

// ---- build bf16 normalized adjacency: an[k][i][j] = di*(A+I)*dj ----
__global__ void build_an_kernel(const float4* __restrict__ adj,
                                const float* __restrict__ dinv,
                                bf16_t* __restrict__ an)
{
    const size_t idx = (size_t)blockIdx.x * 256 + threadIdx.x;   // over N*N
    const int i = (int)(idx >> 12);
    const int j = (int)(idx & (NN_ - 1));
    const float diag = (i == j) ? 1.0f : 0.0f;
    float4 a = adj[idx];
    const float v[4] = { a.x, a.y, a.z, a.w };
    #pragma unroll
    for (int k = 0; k < KG_; ++k) {
        const float di = dinv[k * NN_ + i];
        const float dj = dinv[k * NN_ + j];
        an[(size_t)k * NN_ * NN_ + idx] = f2bf(di * (v[k] + diag) * dj);
    }
}

// ------------------------- small utility kernels -------------------------
__global__ void cvt_bf16_kernel(const float* __restrict__ in, bf16_t* __restrict__ out, int n)
{
    int i = blockIdx.x * 256 + threadIdx.x;
    if (i < n) out[i] = f2bf(in[i]);
}

// in: [R x Cc] fp32 row-major -> out: [Cc x R] bf16 row-major
__global__ void transpose_cvt_kernel(const float* __restrict__ in, bf16_t* __restrict__ out,
                                     int R, int Cc)
{
    int idx = blockIdx.x * 256 + threadIdx.x;
    if (idx >= R * Cc) return;
    int r = idx / Cc, c = idx - r * Cc;
    out[(size_t)c * R + r] = f2bf(in[idx]);
}

__global__ void zero_u32_kernel(unsigned int* __restrict__ p, int n)
{
    int i = blockIdx.x * 256 + threadIdx.x;
    if (i < n) p[i] = 0u;
}

__global__ void add2_kernel(const float* __restrict__ a, const float* __restrict__ b,
                            float* __restrict__ o, int n)
{
    int i = blockIdx.x * 256 + threadIdx.x;
    if (i < n) o[i] = a[i] + b[i];
}

// ------------- LSTM gates: G=[N,4H] (i,f,g,o); c in-place -------------
__global__ void lstm_gate_kernel(const float* __restrict__ G, float* __restrict__ cbuf,
                                 float* __restrict__ h_slice /* + t*H, row stride K*H */,
                                 bf16_t* __restrict__ hbf)
{
    int idx = blockIdx.x * 256 + threadIdx.x;     // over N*H
    int i = idx >> 7, h = idx & (HH_ - 1);
    const float* g = G + (size_t)i * GH_;
    float gi = g[h], gf = g[HH_ + h], gg = g[2 * HH_ + h], go = g[3 * HH_ + h];
    float c  = sigm(gf) * cbuf[idx] + sigm(gi) * tanhf(gg);
    float hn = sigm(go) * tanhf(c);
    cbuf[idx] = c;
    h_slice[(size_t)i * (KG_ * HH_) + h] = hn;
    hbf[idx] = f2bf(hn);
}

// ---- attention over K, pooling, output linear + softmax. 1 wave / node ----
__launch_bounds__(256)
__global__ void attn_out_kernel(const float* __restrict__ gnn,   // [N,K,H]
                                const float* __restrict__ hf,    // [N,K,H]
                                const float* __restrict__ hb,    // [N,K,H]
                                const float* __restrict__ rw,    // [2H]
                                const float* __restrict__ rb,    // [1]
                                const float* __restrict__ ow,    // [C,H]
                                const float* __restrict__ ob,    // [C]
                                float* __restrict__ out)         // [N,C]
{
    const int lane = threadIdx.x & 31;
    const int wave = threadIdx.x >> 5;
    const int i = blockIdx.x * 8 + wave;

    float a[KG_];
    float mx = -1e30f;
    #pragma unroll
    for (int t = 0; t < KG_; ++t) {
        const float* hfr = hf + ((size_t)i * KG_ + t) * HH_;
        const float* hbr = hb + ((size_t)i * KG_ + t) * HH_;
        float p = 0.f;
        #pragma unroll
        for (int s = 0; s < 4; ++s) {
            int h = lane + 32 * s;
            p += hfr[h] * rw[h] + hbr[h] * rw[HH_ + h];
        }
        for (int off = 16; off > 0; off >>= 1) p += __shfl_xor(p, off, 32);
        a[t] = p + rb[0];
        mx = fmaxf(mx, a[t]);
    }
    float ssum = 0.f;
    #pragma unroll
    for (int t = 0; t < KG_; ++t) { a[t] = __expf(a[t] - mx); ssum += a[t]; }
    const float inv = 1.f / ssum;

    float pooled[4];
    #pragma unroll
    for (int s = 0; s < 4; ++s) {
        int h = lane + 32 * s;
        float acc = 0.f;
        #pragma unroll
        for (int t = 0; t < KG_; ++t) acc += gnn[((size_t)i * KG_ + t) * HH_ + h] * a[t];
        pooled[s] = acc * inv;
    }

    float logit[CC_];
    #pragma unroll
    for (int c = 0; c < CC_; ++c) {
        float p = 0.f;
        #pragma unroll
        for (int s = 0; s < 4; ++s) p += pooled[s] * ow[c * HH_ + lane + 32 * s];
        for (int off = 16; off > 0; off >>= 1) p += __shfl_xor(p, off, 32);
        logit[c] = p + ob[c];
    }
    float lm = -1e30f;
    #pragma unroll
    for (int c = 0; c < CC_; ++c) lm = fmaxf(lm, logit[c]);
    float ls = 0.f;
    #pragma unroll
    for (int c = 0; c < CC_; ++c) { logit[c] = __expf(logit[c] - lm); ls += logit[c]; }
    if (lane < CC_) out[(size_t)i * CC_ + lane] = logit[lane] / ls;
}

// =====================================================================
extern "C" void kernel_launch(void* const* d_in, const int* in_sizes, int n_in,
                              void* d_out, int out_size, void* d_ws, size_t ws_size,
                              hipStream_t stream)
{
    (void)in_sizes; (void)n_in; (void)out_size; (void)ws_size;
    const float* feat  = (const float*)d_in[0];   // [N,F]
    const float* adj   = (const float*)d_in[1];   // [N,N,K]
    const float* w1    = (const float*)d_in[2];   // [K,F,H]
    const float* b1    = (const float*)d_in[3];   // [K,H]
    const float* w2    = (const float*)d_in[4];   // [K,H,H]
    const float* b2    = (const float*)d_in[5];   // [K,H]
    const float* wih_f = (const float*)d_in[6];   // [4H,H]
    const float* whh_f = (const float*)d_in[7];
    const float* bih_f = (const float*)d_in[8];
    const float* bhh_f = (const float*)d_in[9];
    const float* wih_b = (const float*)d_in[10];
    const float* whh_b = (const float*)d_in[11];
    const float* bih_b = (const float*)d_in[12];
    const float* bhh_b = (const float*)d_in[13];
    const float* rw    = (const float*)d_in[14];  // [1,2H]
    const float* rb    = (const float*)d_in[15];  // [1]
    const float* ow    = (const float*)d_in[16];  // [C,H]
    const float* ob    = (const float*)d_in[17];  // [C]
    float* out = (float*)d_out;                   // [N,C]

    // --------- bump allocator over d_ws ---------
    char* wp = (char*)d_ws;
    auto alloc = [&](size_t bytes) -> void* {
        void* r = (void*)wp;
        wp += (bytes + 255) & ~(size_t)255;
        return r;
    };
    const size_t NNsz = (size_t)NN_ * NN_;
    float*  dinv  = (float*)alloc((size_t)KG_ * NN_ * 4);
    bf16_t* anbf  = (bf16_t*)alloc((size_t)KG_ * NNsz * 2);
    bf16_t* xbf   = (bf16_t*)alloc((size_t)NN_ * FF_ * 2);
    bf16_t* w1t   = (bf16_t*)alloc((size_t)FF_ * HH_ * 2);
    bf16_t* w2t   = (bf16_t*)alloc((size_t)HH_ * HH_ * 2);
    float*  xw    = (float*)alloc((size_t)NN_ * HH_ * 4);
    bf16_t* xwt   = (bf16_t*)alloc((size_t)HH_ * NN_ * 2);
    float*  h1    = (float*)alloc((size_t)NN_ * HH_ * 4);
    bf16_t* h1bf  = (bf16_t*)alloc((size_t)NN_ * HH_ * 2);
    float*  h1w   = (float*)alloc((size_t)NN_ * HH_ * 4);
    bf16_t* h1wt  = (bf16_t*)alloc((size_t)HH_ * NN_ * 2);
    float*  gnn   = (float*)alloc((size_t)NN_ * KG_ * HH_ * 4);
    bf16_t* gnnbf = (bf16_t*)alloc((size_t)NN_ * KG_ * HH_ * 2);
    bf16_t* wihbf = (bf16_t*)alloc((size_t)GH_ * HH_ * 2);
    bf16_t* whhbf = (bf16_t*)alloc((size_t)GH_ * HH_ * 2);
    float*  biasc = (float*)alloc((size_t)GH_ * 4);
    float*  G     = (float*)alloc((size_t)NN_ * GH_ * 4);
    float*  cbuf  = (float*)alloc((size_t)NN_ * HH_ * 4);
    bf16_t* hbf   = (bf16_t*)alloc((size_t)NN_ * HH_ * 2);
    float*  hfwd  = (float*)alloc((size_t)NN_ * KG_ * HH_ * 4);
    float*  hbwd  = (float*)alloc((size_t)NN_ * KG_ * HH_ * 4);

    // staged WMMA GEMM: M%128==0, N%64==0, Kd%32==0 (all call sites satisfy)
    auto gemm = [&](const bf16_t* A, int lda, const bf16_t* Bt, int ldb,
                    float* C, int ldc, int M, int Nn, int Kd,
                    const float* bias, int accum) {
        dim3 grid(Nn / 64, M / 128), block(32, 8);
        gemm_wmma_stage<<<grid, block, 0, stream>>>(A, lda, Bt, ldb, C, ldc, Kd, bias, accum);
    };

    // ---------------- adjacency normalization ----------------
    degnorm_kernel<<<NN_, 256, 0, stream>>>((const float4*)adj, dinv);
    build_an_kernel<<<(int)(NNsz / 256), 256, 0, stream>>>((const float4*)adj, dinv, anbf);

    // ---------------- GCN (2 layers) per graph block ----------------
    cvt_bf16_kernel<<<(NN_ * FF_) / 256, 256, 0, stream>>>(feat, xbf, NN_ * FF_);
    for (int k = 0; k < KG_; ++k) {
        const bf16_t* ank = anbf + (size_t)k * NNsz;
        // XW = X @ W1[k]
        transpose_cvt_kernel<<<(FF_ * HH_) / 256, 256, 0, stream>>>(w1 + (size_t)k * FF_ * HH_, w1t, FF_, HH_);
        gemm(xbf, FF_, w1t, FF_, xw, HH_, NN_, HH_, FF_, nullptr, 0);
        // H1 = AN @ XW + b1[k]
        transpose_cvt_kernel<<<(NN_ * HH_) / 256, 256, 0, stream>>>(xw, xwt, NN_, HH_);
        gemm(ank, NN_, xwt, NN_, h1, HH_, NN_, HH_, NN_, b1 + k * HH_, 0);
        // H1W = H1 @ W2[k]
        cvt_bf16_kernel<<<(NN_ * HH_) / 256, 256, 0, stream>>>(h1, h1bf, NN_ * HH_);
        transpose_cvt_kernel<<<(HH_ * HH_) / 256, 256, 0, stream>>>(w2 + (size_t)k * HH_ * HH_, w2t, HH_, HH_);
        gemm(h1bf, HH_, w2t, HH_, h1w, HH_, NN_, HH_, HH_, nullptr, 0);
        // gnn[:,k,:] = AN @ H1W + b2[k]
        transpose_cvt_kernel<<<(NN_ * HH_) / 256, 256, 0, stream>>>(h1w, h1wt, NN_, HH_);
        gemm(ank, NN_, h1wt, NN_, gnn + (size_t)k * HH_, KG_ * HH_, NN_, HH_, NN_, b2 + k * HH_, 0);
    }
    cvt_bf16_kernel<<<(NN_ * KG_ * HH_) / 256, 256, 0, stream>>>(gnn, gnnbf, NN_ * KG_ * HH_);

    // ---------------- bidirectional LSTM over K timesteps ----------------
    for (int dir = 0; dir < 2; ++dir) {
        const float* wih = dir ? wih_b : wih_f;
        const float* whh = dir ? whh_b : whh_f;
        const float* bih = dir ? bih_b : bih_f;
        const float* bhh = dir ? bhh_b : bhh_f;
        float* hdir = dir ? hbwd : hfwd;
        // weights used as x @ W^T: Bt is W itself, just convert
        cvt_bf16_kernel<<<(GH_ * HH_) / 256, 256, 0, stream>>>(wih, wihbf, GH_ * HH_);
        cvt_bf16_kernel<<<(GH_ * HH_) / 256, 256, 0, stream>>>(whh, whhbf, GH_ * HH_);
        add2_kernel<<<(GH_ + 255) / 256, 256, 0, stream>>>(bih, bhh, biasc, GH_);
        // zero initial states (c fp32, h bf16)
        zero_u32_kernel<<<(NN_ * HH_) / 256, 256, 0, stream>>>((unsigned int*)cbuf, NN_ * HH_);
        zero_u32_kernel<<<(NN_ * HH_ / 2) / 256, 256, 0, stream>>>((unsigned int*)hbf, NN_ * HH_ / 2);
        for (int step = 0; step < KG_; ++step) {
            int t = dir ? (KG_ - 1 - step) : step;
            // G = bias + X_t @ Wih^T ;  G += h_prev @ Whh^T
            gemm(gnnbf + (size_t)t * HH_, KG_ * HH_, wihbf, HH_, G, GH_, NN_, GH_, HH_, biasc, 0);
            gemm(hbf, HH_, whhbf, HH_, G, GH_, NN_, GH_, HH_, nullptr, 1);
            lstm_gate_kernel<<<(NN_ * HH_) / 256, 256, 0, stream>>>(G, cbuf, hdir + (size_t)t * HH_, hbf);
        }
    }

    // ---------------- attention + pooling + output softmax ----------------
    attn_out_kernel<<<NN_ / 8, 256, 0, stream>>>(gnn, hfwd, hbwd, rw, rb, ow, ob, out);
}
